// DynamicGraphTransformerSuper_42786464203047
// MI455X (gfx1250) — compile-verified, matrix operand order checked by reading, not run on hardware
//
#include <hip/hip_runtime.h>

typedef __attribute__((ext_vector_type(16))) _Float16 v16h;
typedef __attribute__((ext_vector_type(8)))  float    v8f;

#define B_ 128
#define N_ 100
#define NP_ 112   // N padded to multiple of 16 -> unguarded WMMA tiles
#define H_ 32
#define T_ 100

// ---------- WMMA helpers (CDNA5 v_wmma_f32_16x16x32_f16), UNGUARDED ----------
// A fragment (and B fragment for X @ W^T, since B[k][n] = W[n][k]):
// lane indexes the M (or N) row; the 16 f16 elements index K per the ISA layout
//   K(j, lane) = (j/8)*16 + (lane/16)*8 + (j%8)
// -> two contiguous 8-float runs per lane: [cb, cb+8) and [cb+16, cb+24)
__device__ __forceinline__ v16h wmma_frag16u(const float* M, int ld, int r0, int c0) {
  int lane = threadIdx.x & 31;
  int r  = r0 + (lane & 15);
  int cb = c0 + ((lane >> 4) << 3);
  const float* p0 = M + r * ld + cb;
  const float* p1 = p0 + 16;
  v16h f;
#pragma unroll
  for (int j = 0; j < 8; ++j) {
    f[j]     = (_Float16)p0[j];
    f[j + 8] = (_Float16)p1[j];
  }
  return f;
}

// C/D fragment: VGPR j holds M = j + 8*(lane/16), N = lane%16
__device__ __forceinline__ void wmma_store16u(float* M, int ld, int r0, int c0, v8f acc) {
  int lane = threadIdx.x & 31;
  int n  = c0 + (lane & 15);
  int rb = r0 + ((lane >> 4) << 3);
  float* p = M + rb * ld + n;
#pragma unroll
  for (int j = 0; j < 8; ++j) p[j * ld] = acc[j];
}

__device__ __forceinline__ v8f wmma_f16(v16h a, v16h b, v8f c) {
  // 8 args: (neg_a, A, neg_b, B, c_mod, C, reuse_a, reuse_b)
  return __builtin_amdgcn_wmma_f32_16x16x32_f16(false, a, false, b, (short)0, c,
                                                false, false);
}

// ---------- wave32 + 4-wave block reductions (blockDim.x == 128) ----------
__device__ __forceinline__ float wave_red_sum(float v) {
#pragma unroll
  for (int m = 16; m > 0; m >>= 1) v += __shfl_xor(v, m, 32);
  return v;
}
__device__ __forceinline__ float wave_red_max(float v) {
#pragma unroll
  for (int m = 16; m > 0; m >>= 1) v = fmaxf(v, __shfl_xor(v, m, 32));
  return v;
}
__device__ __forceinline__ float block_sum(float v, float* part) {
  int wave = threadIdx.x >> 5, lane = threadIdx.x & 31;
  v = wave_red_sum(v);
  if (lane == 0) part[wave] = v;
  __syncthreads();
  float r = part[0] + part[1] + part[2] + part[3];
  __syncthreads();
  return r;
}
__device__ __forceinline__ float block_max(float v, float* part) {
  int wave = threadIdx.x >> 5, lane = threadIdx.x & 31;
  v = wave_red_max(v);
  if (lane == 0) part[wave] = v;
  __syncthreads();
  float r = fmaxf(fmaxf(part[0], part[1]), fmaxf(part[2], part[3]));
  __syncthreads();
  return r;
}
// argmax of (value, index), ties -> smaller index (matches jnp.argmax)
__device__ __forceinline__ void block_argmax(float v, int idx, float* partv, int* parti,
                                             float* ov, int* oi) {
  int wave = threadIdx.x >> 5, lane = threadIdx.x & 31;
#pragma unroll
  for (int m = 16; m > 0; m >>= 1) {
    float v2 = __shfl_xor(v, m, 32);
    int   i2 = __shfl_xor(idx, m, 32);
    if (v2 > v || (v2 == v && i2 < idx)) { v = v2; idx = i2; }
  }
  if (lane == 0) { partv[wave] = v; parti[wave] = idx; }
  __syncthreads();
  float bv = partv[0]; int bi = parti[0];
#pragma unroll
  for (int w = 1; w < 4; ++w) {
    float v2 = partv[w]; int i2 = parti[w];
    if (v2 > bv || (v2 == bv && i2 < bi)) { bv = v2; bi = i2; }
  }
  __syncthreads();
  *ov = bv; *oi = bi;
}

// ---------- encoder: one block per batch element ----------
__global__ __launch_bounds__(128)
void encoder_kernel(const float* __restrict__ nf, const float* __restrict__ emb_w,
                    const float* __restrict__ emb_b, const float* __restrict__ qkv_w,
                    const float* __restrict__ qkv_b, const float* __restrict__ out_w,
                    const float* __restrict__ out_b, const float* __restrict__ ln_w,
                    const float* __restrict__ ln_b, float* __restrict__ emb_g) {
  __shared__ float X[NP_ * H_];    // residual stream (padded rows zeroed)
  __shared__ float QKV[NP_ * 96];  // q|k|v; q-slot reused for att out, k-slot for proj
  int b = blockIdx.x, tid = threadIdx.x, wave = tid >> 5;
  const float* nfb = nf + b * N_ * 3;

  for (int idx = tid; idx < N_ * H_; idx += 128) {
    int n = idx >> 5, h = idx & 31;
    float a = emb_b[h];
    for (int k = 0; k < 3; ++k) a += nfb[n * 3 + k] * emb_w[h * 3 + k];
    X[idx] = a;
  }
  for (int idx = N_ * H_ + tid; idx < NP_ * H_; idx += 128) X[idx] = 0.0f;  // pad rows
  __syncthreads();

  for (int l = 0; l < 2; ++l) {
    // ---- qkv projection: [112,32] @ [96,32]^T, K=32 exact ----
    const float* qw = qkv_w + l * 96 * H_;
    v16h bq[6];
#pragma unroll
    for (int nt = 0; nt < 6; ++nt) bq[nt] = wmma_frag16u(qw, H_, nt * 16, 0);
    for (int mt = wave; mt < 7; mt += 4) {
      v16h a = wmma_frag16u(X, H_, mt * 16, 0);
#pragma unroll
      for (int nt = 0; nt < 6; ++nt) {
        v8f acc = {};
        acc = wmma_f16(a, bq[nt], acc);
        wmma_store16u(QKV, 96, mt * 16, nt * 16, acc);
      }
    }
    __syncthreads();
    for (int idx = tid; idx < N_ * 96; idx += 128) QKV[idx] += qkv_b[l * 96 + idx % 96];
    __syncthreads();

    // ---- attention: results to registers, then write into the q-slot ----
    float oa[16], ob[16];
    {
      int pair = tid;  // always < 200
      int h = pair / N_, i = pair % N_;
      const float* qi = QKV + i * 96 + h * 16;
      float mx = -3.4e38f;
      for (int j = 0; j < N_; ++j) {
        const float* kj = QKV + j * 96 + 32 + h * 16;
        float s = 0.0f;
        for (int d = 0; d < 16; ++d) s += qi[d] * kj[d];
        mx = fmaxf(mx, s * 0.25f);
      }
      float sum = 0.0f;
      for (int d = 0; d < 16; ++d) oa[d] = 0.0f;
      for (int j = 0; j < N_; ++j) {
        const float* kj = QKV + j * 96 + 32 + h * 16;
        const float* vj = QKV + j * 96 + 64 + h * 16;
        float s = 0.0f;
        for (int d = 0; d < 16; ++d) s += qi[d] * kj[d];
        float e = __expf(s * 0.25f - mx);
        sum += e;
        for (int d = 0; d < 16; ++d) oa[d] += e * vj[d];
      }
      float inv = 1.0f / sum;
      for (int d = 0; d < 16; ++d) oa[d] *= inv;
    }
    if (tid < 72) {
      int pair = tid + 128;
      int h = pair / N_, i = pair % N_;
      const float* qi = QKV + i * 96 + h * 16;
      float mx = -3.4e38f;
      for (int j = 0; j < N_; ++j) {
        const float* kj = QKV + j * 96 + 32 + h * 16;
        float s = 0.0f;
        for (int d = 0; d < 16; ++d) s += qi[d] * kj[d];
        mx = fmaxf(mx, s * 0.25f);
      }
      float sum = 0.0f;
      for (int d = 0; d < 16; ++d) ob[d] = 0.0f;
      for (int j = 0; j < N_; ++j) {
        const float* kj = QKV + j * 96 + 32 + h * 16;
        const float* vj = QKV + j * 96 + 64 + h * 16;
        float s = 0.0f;
        for (int d = 0; d < 16; ++d) s += qi[d] * kj[d];
        float e = __expf(s * 0.25f - mx);
        sum += e;
        for (int d = 0; d < 16; ++d) ob[d] += e * vj[d];
      }
      float inv = 1.0f / sum;
      for (int d = 0; d < 16; ++d) ob[d] *= inv;
    }
    __syncthreads();  // everyone done reading q/k/v
    {
      int h = tid / N_, i = tid % N_;
      for (int d = 0; d < 16; ++d) QKV[i * 96 + h * 16 + d] = oa[d];
      if (tid < 72) {
        int p2 = tid + 128, h2 = p2 / N_, i2 = p2 % N_;
        for (int d = 0; d < 16; ++d) QKV[i2 * 96 + h2 * 16 + d] = ob[d];
      }
    }
    __syncthreads();

    // ---- out projection: att(q-slot) @ out_w^T -> k-slot (cols 32..63) ----
    const float* ow = out_w + l * H_ * H_;
    v16h bo[2];
#pragma unroll
    for (int nt = 0; nt < 2; ++nt) bo[nt] = wmma_frag16u(ow, H_, nt * 16, 0);
    for (int mt = wave; mt < 7; mt += 4) {
      v16h a = wmma_frag16u(QKV, 96, mt * 16, 0);
#pragma unroll
      for (int nt = 0; nt < 2; ++nt) {
        v8f acc = {};
        acc = wmma_f16(a, bo[nt], acc);
        wmma_store16u(QKV, 96, mt * 16, 32 + nt * 16, acc);
      }
    }
    __syncthreads();

    // ---- residual + bias + layernorm -> X ----
    for (int n = tid; n < N_; n += 128) {
      float y[H_], mu = 0.0f;
      for (int h = 0; h < H_; ++h) {
        y[h] = X[n * H_ + h] + QKV[n * 96 + 32 + h] + out_b[l * H_ + h];
        mu += y[h];
      }
      mu *= (1.0f / H_);
      float var = 0.0f;
      for (int h = 0; h < H_; ++h) { float d = y[h] - mu; var += d * d; }
      var *= (1.0f / H_);
      float inv = rsqrtf(var + 1e-5f);
      for (int h = 0; h < H_; ++h)
        X[n * H_ + h] = (y[h] - mu) * inv * ln_w[l * H_ + h] + ln_b[l * H_ + h];
    }
    __syncthreads();
  }
  for (int idx = tid; idx < N_ * H_; idx += 128) emb_g[b * N_ * H_ + idx] = X[idx];
}

// ---------- decoder: one block per batch, all 100 steps in-kernel ----------
__global__ __launch_bounds__(128)
void decoder_kernel(const float* __restrict__ emb_g, const float* __restrict__ demands,
                    const float* __restrict__ caps, const float* __restrict__ se_w,
                    const float* __restrict__ se_b, const float* __restrict__ du_w,
                    const float* __restrict__ du_b, const float* __restrict__ p1_w,
                    const float* __restrict__ p1_b, const float* __restrict__ p2_w,
                    const float* __restrict__ p2_b, const float* __restrict__ p3_w,
                    const float* __restrict__ p3_b, float* __restrict__ acts_out,
                    float* __restrict__ sel_g, float* __restrict__ ent_g,
                    int* __restrict__ done1_g) {
  __shared__ float EMB[NP_ * H_], UE[NP_ * H_], H1[NP_ * H_], H2[NP_ * 16];
  __shared__ float partv[4]; __shared__ int parti[4];
  __shared__ float s_scores[N_];
  __shared__ unsigned char vis[N_], s_mask[N_];
  __shared__ float s_st[8], s_stc[32], s_gc[32], s_gcc[32];
  __shared__ float s_rem; __shared__ int s_cur, s_done, s_flag, s_cnt;
  int b = blockIdx.x, tid = threadIdx.x, wave = tid >> 5;
  float cap = caps[b];
  const float* dem = demands + b * N_;

  for (int idx = tid; idx < N_ * H_; idx += 128) EMB[idx] = emb_g[b * N_ * H_ + idx];
  for (int idx = N_ * H_ + tid; idx < NP_ * H_; idx += 128) EMB[idx] = 0.0f;
  if (tid < N_) vis[tid] = 0;
  if (tid == 0) { s_rem = cap; s_cur = 0; s_done = 0; }

  // loop-invariant B fragments (weights), K = 32 exact for all
  v16h bdu0 = wmma_frag16u(du_w, 40, 0, 0);
  v16h bdu1 = wmma_frag16u(du_w, 40, 16, 0);
  v16h bp10 = wmma_frag16u(p1_w, 64, 0, 0);
  v16h bp11 = wmma_frag16u(p1_w, 64, 16, 0);
  v16h bp2  = wmma_frag16u(p2_w, H_, 0, 0);
  __syncthreads();

  for (int t = 0; t < T_; ++t) {
    // ---- done update #1: all customers visited and at depot ----
    if (tid == 0) s_flag = 1;
    __syncthreads();
    if (tid >= 1 && tid < N_ && !vis[tid]) s_flag = 0;
    __syncthreads();
    if (tid == 0) {
      if (s_flag && s_cur == 0) s_done = 1;
      done1_g[t * B_ + b] = s_done;   // "active" gating uses done after update #1
      s_cnt = 0;
    }
    __syncthreads();
    if (tid < N_ && vis[tid]) atomicAdd(&s_cnt, 1);
    __syncthreads();

    // ---- state encoding + broadcast term of the gated update ----
    if (tid < 8) {
      float cu = (cap - s_rem) / cap;
      float vc = (float)s_cnt / (float)N_;
      s_st[tid] = se_w[tid * 2] * cu + se_w[tid * 2 + 1] * vc + se_b[tid];
    }
    __syncthreads();
    if (tid < 32) {
      float a = du_b[tid];
      for (int e = 0; e < 8; ++e) a += du_w[tid * 40 + 32 + e] * s_st[e];
      s_stc[tid] = a;
    }
    __syncthreads();

    // ---- ue = emb + 0.3*(emb @ du_w[:, :32]^T + stc) ----
    for (int mt = wave; mt < 7; mt += 4) {
      v16h a = wmma_frag16u(EMB, H_, mt * 16, 0);
      v8f acc0 = {}; acc0 = wmma_f16(a, bdu0, acc0);
      wmma_store16u(UE, H_, mt * 16, 0, acc0);
      v8f acc1 = {}; acc1 = wmma_f16(a, bdu1, acc1);
      wmma_store16u(UE, H_, mt * 16, 16, acc1);
    }
    __syncthreads();
    for (int idx = tid; idx < N_ * H_; idx += 128)
      UE[idx] = EMB[idx] + 0.3f * (UE[idx] + s_stc[idx & 31]);
    __syncthreads();

    // ---- global context + broadcast term of p1 ----
    if (tid < 32) {
      float s = 0.0f;
      for (int n = 0; n < N_; ++n) s += UE[n * H_ + tid];
      s_gc[tid] = s * 0.01f;
    }
    __syncthreads();
    if (tid < 32) {
      float a = p1_b[tid];
      for (int k = 0; k < 32; ++k) a += p1_w[tid * 64 + 32 + k] * s_gc[k];
      s_gcc[tid] = a;
    }
    __syncthreads();

    // ---- h1 = relu(ue @ p1_w[:, :32]^T + gcc) ----
    for (int mt = wave; mt < 7; mt += 4) {
      v16h a = wmma_frag16u(UE, H_, mt * 16, 0);
      v8f acc0 = {}; acc0 = wmma_f16(a, bp10, acc0);
      wmma_store16u(H1, H_, mt * 16, 0, acc0);
      v8f acc1 = {}; acc1 = wmma_f16(a, bp11, acc1);
      wmma_store16u(H1, H_, mt * 16, 16, acc1);
    }
    __syncthreads();
    for (int idx = tid; idx < N_ * H_; idx += 128)
      H1[idx] = fmaxf(H1[idx] + s_gcc[idx & 31], 0.0f);
    __syncthreads();

    // ---- h2 = relu(h1 @ p2_w^T + b2) ----
    for (int mt = wave; mt < 7; mt += 4) {
      v16h a = wmma_frag16u(H1, H_, mt * 16, 0);
      v8f acc = {}; acc = wmma_f16(a, bp2, acc);
      wmma_store16u(H2, 16, mt * 16, 0, acc);
    }
    __syncthreads();
    for (int idx = tid; idx < N_ * 16; idx += 128)
      H2[idx] = fmaxf(H2[idx] + p2_b[idx & 15], 0.0f);
    __syncthreads();

    // ---- scores ----
    if (tid < N_) {
      float s = p3_b[0];
      for (int k = 0; k < 16; ++k) s += H2[tid * 16 + k] * p3_w[k];
      s_scores[tid] = s;
    }
    if (tid == 0) s_cnt = 0;
    __syncthreads();

    // ---- masking ----
    if (tid < N_) {
      int m = (vis[tid] || (dem[tid] > s_rem)) ? 1 : 0;
      if (tid == 0 && s_cur == 0) m = 1;
      s_mask[tid] = (unsigned char)m;
    }
    __syncthreads();
    if (tid < N_) atomicAdd(&s_cnt, (int)s_mask[tid]);
    __syncthreads();
    if (tid == 0) {
      int at_dep = (s_cur == 0);
      int all_m  = (s_cnt == N_);
      if (all_m && !at_dep) s_mask[0] = 0;
      if (all_m && at_dep) s_done = 1;   // done update #2
    }
    __syncthreads();

    // ---- masked softmax, entropy, argmax(logp) ----
    float sc = (tid < N_) ? (s_mask[tid] ? -1e9f : s_scores[tid]) : -3.4e38f;
    float smax = block_max(sc, partv);
    float e = (tid < N_) ? __expf(sc - smax) : 0.0f;
    float ssum = block_sum(e, partv);
    float p  = e / ssum;
    float lg = __logf(p + 1e-10f);
    float entc = (tid < N_) ? (-p * lg) : 0.0f;
    float ent = block_sum(entc, partv);
    float mval; int midx;
    block_argmax((tid < N_) ? lg : -3.4e38f, tid, partv, parti, &mval, &midx);

    if (tid == 0) {
      int act = s_done ? 0 : midx;
      acts_out[t * B_ + b] = (float)act;
      sel_g[t * B_ + b] = s_done ? 0.0f : mval;
      ent_g[t * B_ + b] = ent;
      if (!s_done) {
        if (act == 0) s_rem = cap;
        else { s_rem -= dem[act]; vis[act] = 1; }
        s_cur = act;
      }
    }
    __syncthreads();
  }
}

// ---------- finalize: apply cross-batch "active" gating ----------
__global__ __launch_bounds__(128)
void finalize_kernel(const float* __restrict__ sel_g, const float* __restrict__ ent_g,
                     const int* __restrict__ done1_g, float* __restrict__ lp_out,
                     float* __restrict__ ent_out) {
  int b = threadIdx.x;  // 128 threads, one per batch element
  __shared__ int cnt;
  float lp = 0.0f, ent = 0.0f;
  for (int t = 0; t < T_; ++t) {
    if (b == 0) cnt = 0;
    __syncthreads();
    atomicAdd(&cnt, done1_g[t * B_ + b]);
    __syncthreads();
    int c = cnt;
    __syncthreads();
    if (c < B_) {  // active = not all done (after update #1)
      lp  += sel_g[t * B_ + b];
      ent += ent_g[t * B_ + b];
    }
  }
  lp_out[b] = lp;
  ent_out[b] = ent;
}

extern "C" void kernel_launch(void* const* d_in, const int* in_sizes, int n_in,
                              void* d_out, int out_size, void* d_ws, size_t ws_size,
                              hipStream_t stream) {
  const float* nf    = (const float*)d_in[0];
  const float* dem   = (const float*)d_in[1];
  const float* caps  = (const float*)d_in[2];
  const float* emb_w = (const float*)d_in[3];
  const float* emb_b = (const float*)d_in[4];
  const float* qkv_w = (const float*)d_in[5];
  const float* qkv_b = (const float*)d_in[6];
  const float* out_w = (const float*)d_in[7];
  const float* out_b = (const float*)d_in[8];
  const float* ln_w  = (const float*)d_in[9];
  const float* ln_b  = (const float*)d_in[10];
  const float* se_w  = (const float*)d_in[11];
  const float* se_b  = (const float*)d_in[12];
  const float* du_w  = (const float*)d_in[13];
  const float* du_b  = (const float*)d_in[14];
  const float* p1_w  = (const float*)d_in[15];
  const float* p1_b  = (const float*)d_in[16];
  const float* p2_w  = (const float*)d_in[17];
  const float* p2_b  = (const float*)d_in[18];
  const float* p3_w  = (const float*)d_in[19];
  const float* p3_b  = (const float*)d_in[20];

  float* ws = (float*)d_ws;
  float* emb_g   = ws;                      // 128*100*32 = 409600 floats
  float* sel_g   = ws + 409600;             // 100*128    = 12800
  float* ent_g   = ws + 422400;             // 12800
  int*   done1_g = (int*)(ws + 435200);     // 12800

  float* out  = (float*)d_out;
  float* acts = out;            // [100,128]
  float* lp   = out + 12800;    // [128]
  float* ento = out + 12928;    // [128]

  encoder_kernel<<<B_, 128, 0, stream>>>(nf, emb_w, emb_b, qkv_w, qkv_b, out_w,
                                         out_b, ln_w, ln_b, emb_g);
  decoder_kernel<<<B_, 128, 0, stream>>>(emb_g, dem, caps, se_w, se_b, du_w, du_b,
                                         p1_w, p1_b, p2_w, p2_b, p3_w, p3_b, acts,
                                         sel_g, ent_g, done1_g);
  finalize_kernel<<<1, 128, 0, stream>>>(sel_g, ent_g, done1_g, lp, ento);
}
